// Physics_Attention_Structured_Mesh_3D_12506944766454
// MI455X (gfx1250) — matrix operand order repaired and checked by reading, not compiled, not run on hardware
//
#include <hip/hip_runtime.h>
#include <hip/hip_bf16.h>
#include <stdint.h>

typedef _Float16 half_t;
typedef __attribute__((ext_vector_type(16))) _Float16 v16h;
typedef __attribute__((ext_vector_type(8)))  _Float16 v8h;
typedef __attribute__((ext_vector_type(8)))  float    v8f;

#define DEV __device__ __forceinline__
#define PADL 34
#define PADN (34 * 34 * 34)   // 39304 padded tokens per batch

// ---------------- small vector helpers ----------------
DEV v16h ld16(const half_t* p0, const half_t* p1) {
    v8h a = *(const v8h*)p0;
    v8h b = *(const v8h*)p1;
    v16h r;
#pragma unroll
    for (int i = 0; i < 8; i++) { r[i] = a[i]; r[8 + i] = b[i]; }
    return r;
}
DEV v16h zero16() {
    v16h r;
#pragma unroll
    for (int i = 0; i < 16; i++) r[i] = (half_t)0;
    return r;
}
DEV v8f zero8() {
    v8f r;
#pragma unroll
    for (int i = 0; i < 8; i++) r[i] = 0.0f;
    return r;
}
DEV v8f wmma16(v16h a, v16h b, v8f c) {
    return __builtin_amdgcn_wmma_f32_16x16x32_f16(false, a, false, b, (short)0, c, false, false);
}

// ---------------- Erwin block params ----------------
struct BlockP {
    const float *ln1_g, *ln1_b, *qkv_w, *qkv_b, *proj_w, *proj_b,
                *pe_w, *pe_b, *sigma, *ln2_g, *ln2_b, *w12, *b12, *w3, *b3;
};

// ======================================================
// conversions
// ======================================================
__global__ void k_f32_to_f16(const float* __restrict__ src, half_t* __restrict__ dst, long n) {
    long i = (long)blockIdx.x * 256 + threadIdx.x;
    if (i < n) dst[i] = (half_t)src[i];
}

// padded f16 input: xp[b][34*34*34][128], zero halo -> branch-free conv taps
__global__ void k_pad_x(const float* __restrict__ X, half_t* __restrict__ xp) {
    long i = (long)blockIdx.x * 256 + threadIdx.x;
    if (i >= (long)4 * PADN * 128) return;
    int ic = (int)(i & 127);
    long rest = i >> 7;
    int p  = (int)(rest % PADN);
    int b_ = (int)(rest / PADN);
    int pd = p % 34, pw = (p / 34) % 34, ph = p / 1156;
    half_t v = (half_t)0;
    if (pd >= 1 && pd <= 32 && pw >= 1 && pw <= 32 && ph >= 1 && ph <= 32) {
        long src = ((long)b_ * 32768 + (long)(ph - 1) * 1024 + (pw - 1) * 32 + (pd - 1)) * 128 + ic;
        v = (half_t)X[src];
    }
    xp[i] = v;
}

// wh[t][oc][ic], oc 0..511 from Wfx, 512..1023 from Wx ; W layout [O][I][3][3][3]
__global__ void k_build_wh(const float* __restrict__ Wfx, const float* __restrict__ Wx,
                           half_t* __restrict__ wh) {
    long i = (long)blockIdx.x * 256 + threadIdx.x;
    if (i >= (long)27 * 1024 * 128) return;
    int ic = (int)(i & 127);
    int oc = (int)((i >> 7) & 1023);
    int tp = (int)(i >> 17);
    float w = (oc < 512) ? Wfx[((long)(oc * 128 + ic)) * 27 + tp]
                         : Wx [((long)((oc - 512) * 128 + ic)) * 27 + tp];
    wh[i] = (half_t)w;
}

// ======================================================
// fused dual 3x3x3 conv as implicit GEMM (WMMA)
// grid: 16384 blocks x 256 threads; block = 8 waves sharing one 64-channel
// group (B tile staged in LDS, double buffered), each wave owns 16 rows.
// ======================================================
__global__ void __launch_bounds__(256)
k_conv(const half_t* __restrict__ xp, const half_t* __restrict__ wh,
       const float* __restrict__ bfx, const float* __restrict__ bx,
       half_t* __restrict__ yh) {
    __shared__ half_t bsm[2][64 * 136];          // 136-half rows: bank-conflict padding

    int tid  = threadIdx.x;
    int wv   = tid >> 5;
    int lane = tid & 31;
    int r  = lane & 15;
    int hi = lane >> 4;
    int mg = blockIdx.x >> 4;                    // 1024 groups of 128 rows
    int ng = blockIdx.x & 15;
    int n0 = ng * 64;
    long m0 = ((long)mg * 8 + wv) * 16;          // this wave's 16 rows (global over B*N)
    int b_   = (int)(m0 >> 15);
    int rt   = (int)(m0 & 32767) + r;
    int d0 = rt & 31, w0 = (rt >> 5) & 31, h0 = rt >> 10;
    int p0 = ((h0 + 1) * PADL + (w0 + 1)) * PADL + (d0 + 1);
    long abase0 = ((long)b_ * PADN + p0) * 128;

    // staging assignment: thread copies 64 bytes (one quarter-row) per tap
    int srow = tid >> 2;                          // 0..63 (local oc)
    int sseg = (tid & 3) * 32;                    // half offset within 128-half row

    // preload tap 0 weights into LDS buffer 0
    {
        const half_t* src = wh + (long)n0 * 128 + (long)srow * 128 + sseg;
        v8h t0 = ((const v8h*)src)[0];
        v8h t1 = ((const v8h*)src)[1];
        v8h t2 = ((const v8h*)src)[2];
        v8h t3 = ((const v8h*)src)[3];
        half_t* dst = &bsm[0][srow * 136 + sseg];
        ((v8h*)dst)[0] = t0; ((v8h*)dst)[1] = t1;
        ((v8h*)dst)[2] = t2; ((v8h*)dst)[3] = t3;
    }

    v8f acc[4];
#pragma unroll
    for (int j2 = 0; j2 < 4; j2++) acc[j2] = zero8();

    // preload A for (t=0, kc=0): tap offset (-1,-1,-1) -> -1191 tokens
    v16h a_cur;
    {
        long ab = abase0 + (long)(-1191) * 128 + hi * 8;
        a_cur = ld16(xp + ab, xp + ab + 16);
    }
    __syncthreads();

    for (int t = 0; t < 27; t++) {
        half_t* bufc = bsm[t & 1];
        // issue global loads for next tap's B tile (consumed after compute)
        v8h s0, s1, s2, s3;
        if (t < 26) {
            const half_t* src = wh + (long)(t + 1) * 131072 + (long)n0 * 128
                              + (long)srow * 128 + sseg;
            s0 = ((const v8h*)src)[0];
            s1 = ((const v8h*)src)[1];
            s2 = ((const v8h*)src)[2];
            s3 = ((const v8h*)src)[3];
            if (t < 25)
                __builtin_prefetch(wh + (long)(t + 2) * 131072 + (long)n0 * 128 + tid * 32, 0, 1);
        }
#pragma unroll
        for (int kc = 0; kc < 4; kc++) {
            // prefetch A for next (t,kc) step into alternate registers
            v16h a_next = zero16();
            {
                int tn  = (kc == 3) ? t + 1 : t;
                int kcn = (kc + 1) & 3;
                if (tn < 27) {
                    int dh = tn / 9 - 1, dw = (tn / 3) % 3 - 1, dz = tn % 3 - 1;
                    long ab = abase0 + (long)(dh * 1156 + dw * 34 + dz) * 128 + kcn * 32 + hi * 8;
                    a_next = ld16(xp + ab, xp + ab + 16);
                }
            }
#pragma unroll
            for (int j2 = 0; j2 < 4; j2++) {
                const half_t* bp = bufc + (j2 * 16 + r) * 136 + kc * 32 + hi * 8;
                v16h bm = ld16(bp, bp + 16);
                acc[j2] = wmma16(a_cur, bm, acc[j2]);
            }
            a_cur = a_next;
        }
        if (t < 26) {
            half_t* dst = &bsm[(t + 1) & 1][srow * 136 + sseg];
            ((v8h*)dst)[0] = s0; ((v8h*)dst)[1] = s1;
            ((v8h*)dst)[2] = s2; ((v8h*)dst)[3] = s3;
        }
        __syncthreads();
    }

#pragma unroll
    for (int j2 = 0; j2 < 4; j2++) {
        int oc = n0 + j2 * 16 + r;
        float bias = (oc < 512) ? bfx[oc] : bx[oc - 512];
#pragma unroll
        for (int j = 0; j < 8; j++) {
            long m = m0 + j + hi * 8;
            yh[m * 1024 + oc] = (half_t)(acc[j2][j] + bias);
        }
    }
}

// ======================================================
// slice weights: sw = softmax((x_mid @ Wslice^T + b)/temp) over G=32
// ======================================================
__global__ void __launch_bounds__(256)
k_slice(const half_t* __restrict__ yh, const float* __restrict__ Wslice,
        const float* __restrict__ bslice, const float* __restrict__ temperature,
        half_t* __restrict__ swh, float* __restrict__ swpart) {
    int bh  = blockIdx.x >> 8;
    int blk = blockIdx.x & 255;
    int wave = threadIdx.x >> 5;
    int lane = threadIdx.x & 31;
    int b_ = bh >> 3, h_ = bh & 7;
    float tmp = temperature[h_];
    float temp = fminf(fmaxf(tmp, 0.1f), 5.0f);
    float wr[64];
#pragma unroll
    for (int c = 0; c < 64; c++) wr[c] = Wslice[lane * 64 + c];
    float bsl = bslice[lane];
    float lsum = 0.0f;
    for (int rr = 0; rr < 16; rr++) {
        int n = blk * 128 + wave * 16 + rr;
        const half_t* xm = yh + ((long)b_ * 32768 + n) * 1024 + 512 + h_ * 64;
        float r0 = (float)xm[lane];
        float r1 = (float)xm[32 + lane];
        float logit = 0.0f;
#pragma unroll
        for (int c = 0; c < 32; c++) {
            logit += __shfl(r0, c, 32) * wr[c];
            logit += __shfl(r1, c, 32) * wr[32 + c];
        }
        logit = (logit + bsl) / temp;
        float mx = logit;
        for (int off = 16; off; off >>= 1) mx = fmaxf(mx, __shfl_xor(mx, off, 32));
        float e = __expf(logit - mx);
        float s = e;
        for (int off = 16; off; off >>= 1) s += __shfl_xor(s, off, 32);
        float swv = e / s;
        swh[(((long)bh * 32768 + n) * 32) + lane] = (half_t)swv;
        lsum += swv;
    }
    __shared__ float part[8][32];
    part[wave][lane] = lsum;
    __syncthreads();
    if (wave == 0) {
        float t = 0.0f;
        for (int w = 0; w < 8; w++) t += part[w][lane];
        swpart[(long)blockIdx.x * 32 + lane] = t;
    }
}

__global__ void k_snorm_reduce(const float* __restrict__ part, float* __restrict__ snorm) {
    int o = blockIdx.x * 256 + threadIdx.x;
    if (o >= 1024) return;
    int bh = o >> 5, g = o & 31;
    float s = 0.0f;
    for (int b2 = 0; b2 < 256; b2++) s += part[((long)(bh * 256 + b2)) * 32 + g];
    snorm[o] = s;
}

// tok partials: tokpart[(bh*16+chunk)][g*64+c] = sum over 2048 n of fx*sw
__global__ void __launch_bounds__(256)
k_tok_part(const half_t* __restrict__ yh, const half_t* __restrict__ swh,
           float* __restrict__ tokpart) {
    int bh = blockIdx.x >> 4, chunk = blockIdx.x & 15;
    int tid = threadIdx.x;
    int g = tid >> 3;
    int c0 = (tid & 7) * 8;
    int b_ = bh >> 3, h_ = bh & 7;
    float acc[8];
#pragma unroll
    for (int j = 0; j < 8; j++) acc[j] = 0.0f;
    for (int nn = 0; nn < 2048; nn++) {
        int n = chunk * 2048 + nn;
        float swv = (float)swh[((long)bh * 32768 + n) * 32 + g];
        const half_t* fx = yh + ((long)b_ * 32768 + n) * 1024 + h_ * 64 + c0;
        v8h f = *(const v8h*)fx;
#pragma unroll
        for (int j = 0; j < 8; j++) acc[j] += swv * (float)f[j];
    }
#pragma unroll
    for (int j = 0; j < 8; j++)
        tokpart[(long)blockIdx.x * 2048 + tid * 8 + j] = acc[j];
}

__global__ void k_tok_reduce(const float* __restrict__ tokpart, const float* __restrict__ snorm,
                             float* __restrict__ tokens0) {
    int i = blockIdx.x * 256 + threadIdx.x;
    if (i >= 65536) return;
    int bh = i >> 11, gc = i & 2047;
    float s = 0.0f;
    for (int ch = 0; ch < 16; ch++) s += tokpart[((long)(bh * 16 + ch)) * 2048 + gc];
    tokens0[i] = s / (snorm[i >> 6] + 1e-5f);
}

// ======================================================
// Erwin small kernels
// ======================================================
DEV int mpart(int v) {
    v = (v | (v << 16)) & 0x030000FF;
    v = (v | (v << 8))  & 0x0300F00F;
    v = (v | (v << 4))  & 0x030C30C3;
    v = (v | (v << 2))  & 0x09249249;
    return v;
}

__global__ void k_morton(const float* __restrict__ pos, int* __restrict__ idx,
                         float* __restrict__ ps0) {
    int g = blockIdx.x, i = threadIdx.x;    // 32 groups x 32
    int t = g * 32 + i;
    int q[3];
    for (int a = 0; a < 3; a++) {
        int v = (int)(pos[t * 3 + a] * 1024.0f);
        q[a] = v < 0 ? 0 : (v > 1023 ? 1023 : v);
    }
    int code = mpart(q[0]) | (mpart(q[1]) << 1) | (mpart(q[2]) << 2);
    __shared__ int codes[32];
    codes[i] = code;
    __syncthreads();
    int rank = 0;
    for (int j = 0; j < 32; j++)
        rank += (codes[j] < code) || (codes[j] == code && j < i);
    idx[g * 32 + rank] = t;
    for (int a = 0; a < 3; a++) ps0[(g * 32 + rank) * 3 + a] = pos[t * 3 + a];
}

// out[m,n] = (res? res : 0) + b[n] + in[rowidx? rowidx[m]:m, :] . W[n, :]
__global__ void k_gemm(const float* in, const float* W, const float* bias,
                       const float* res, const int* rowidx, float* out,
                       int M, int Nn, int K) {
    long i = (long)blockIdx.x * 256 + threadIdx.x;
    if (i >= (long)M * Nn) return;
    int m = (int)(i / Nn), nn = (int)(i % Nn);
    int row = rowidx ? rowidx[m] : m;
    const float* ip = in + (long)row * K;
    const float* wp = W + (long)nn * K;
    float s = bias ? bias[nn] : 0.0f;
    for (int k = 0; k < K; k++) s += ip[k] * wp[k];
    if (res) s += res[i];
    out[i] = s;
}

__global__ void __launch_bounds__(256)
k_layernorm(const float* __restrict__ in, float* __restrict__ out,
            const float* __restrict__ g, const float* __restrict__ b, int M, int D) {
    int row = blockIdx.x * 8 + (threadIdx.x >> 5);
    int lane = threadIdx.x & 31;
    if (row >= M) return;
    const float* ip = in + (long)row * D;
    int per = D >> 5;                 // 2 or 4
    float vals[4];
    float s = 0.0f, ss = 0.0f;
    for (int t = 0; t < per; t++) {
        float x = ip[lane + t * 32];
        vals[t] = x; s += x; ss += x * x;
    }
    for (int off = 16; off; off >>= 1) {
        s  += __shfl_xor(s,  off, 32);
        ss += __shfl_xor(ss, off, 32);
    }
    float mean = s / D;
    float var  = ss / D - mean * mean;
    float inv  = rsqrtf(var + 1e-5f);
    float* op = out + (long)row * D;
    for (int t = 0; t < per; t++) {
        int c = lane + t * 32;
        op[c] = (vals[t] - mean) * inv * g[c] + b[c];
    }
}

__global__ void k_glu(const float* __restrict__ u, float* __restrict__ gout, int M, int hid) {
    long i = (long)blockIdx.x * 256 + threadIdx.x;
    if (i >= (long)M * hid) return;
    int m = (int)(i / hid), j = (int)(i % hid);
    float a  = u[(long)m * 2 * hid + j];
    float b2 = u[(long)m * 2 * hid + hid + j];
    gout[i] = (a / (1.0f + __expf(-a))) * b2;
}

// ball attention (includes pe, qkv, attn+dist bias, proj, residual add)
__global__ void __launch_bounds__(256)
k_ball(const float* __restrict__ xln, float* __restrict__ x,
       const float* __restrict__ pos, BlockP p, int dim, int heads, int ball) {
    extern __shared__ float s[];
    int ballDim = ball * dim;
    float* sp  = s;                        // ball*3 rel positions
    float* xb  = sp + ball * 3;            // ball*dim
    float* qs  = xb + ballDim;
    float* ks  = qs + ballDim;
    float* vs  = ks + ballDim;
    float* os  = vs + ballDim;
    float* att = os + ballDim;             // heads*ball*ball
    float* ctr = att + heads * ball * ball;

    int tid = threadIdx.x;
    long base = (long)blockIdx.x * ball;
    int hd = dim / heads;

    for (int i = tid; i < ball * 3; i += 256) sp[i] = pos[base * 3 + i];
    __syncthreads();
    if (tid < 3) {
        float m = 0.0f;
        for (int i = 0; i < ball; i++) m += sp[i * 3 + tid];
        ctr[tid] = m / ball;
    }
    __syncthreads();
    for (int i = tid; i < ball * 3; i += 256) sp[i] -= ctr[i % 3];
    __syncthreads();
    for (int e = tid; e < ballDim; e += 256) {
        int i = e / dim, c = e % dim;
        xb[e] = xln[(base + i) * dim + c]
              + sp[i * 3 + 0] * p.pe_w[c * 3 + 0]
              + sp[i * 3 + 1] * p.pe_w[c * 3 + 1]
              + sp[i * 3 + 2] * p.pe_w[c * 3 + 2]
              + p.pe_b[c];
    }
    __syncthreads();
    for (int e = tid; e < 3 * ballDim; e += 256) {
        int which = e / ballDim;
        int e2 = e % ballDim;
        int i = e2 / dim, c = e2 % dim;
        int j = which * dim + c;
        float sa = p.qkv_b[j];
        const float* wr = p.qkv_w + (long)j * dim;
        const float* xr = xb + i * dim;
        for (int cc = 0; cc < dim; cc++) sa += xr[cc] * wr[cc];
        float* dst = (which == 0) ? qs : (which == 1) ? ks : vs;
        dst[e2] = sa;
    }
    __syncthreads();
    float scale = rsqrtf((float)hd);
    for (int e = tid; e < heads * ball * ball; e += 256) {
        int h = e / (ball * ball);
        int rem = e % (ball * ball);
        int i = rem / ball, j = rem % ball;
        float sa = 0.0f;
        const float* qi = qs + i * dim + h * hd;
        const float* kj = ks + j * dim + h * hd;
        for (int cc = 0; cc < hd; cc++) sa += qi[cc] * kj[cc];
        float dx = sp[i * 3] - sp[j * 3];
        float dy = sp[i * 3 + 1] - sp[j * 3 + 1];
        float dz = sp[i * 3 + 2] - sp[j * 3 + 2];
        float dist = sqrtf(dx * dx + dy * dy + dz * dz + 1e-12f);
        att[e] = sa * scale + p.sigma[h] * dist;
    }
    __syncthreads();
    for (int rrow = tid; rrow < heads * ball; rrow += 256) {
        float* ar = att + (long)rrow * ball;
        float mx = -1e30f;
        for (int j = 0; j < ball; j++) mx = fmaxf(mx, ar[j]);
        float ssum = 0.0f;
        for (int j = 0; j < ball; j++) { float e2 = __expf(ar[j] - mx); ar[j] = e2; ssum += e2; }
        float inv = 1.0f / ssum;
        for (int j = 0; j < ball; j++) ar[j] *= inv;
    }
    __syncthreads();
    for (int e = tid; e < ballDim; e += 256) {
        int i = e / dim, c = e % dim;
        int h = c / hd;
        const float* ar = att + ((long)h * ball + i) * ball;
        float sa = 0.0f;
        for (int j = 0; j < ball; j++) sa += ar[j] * vs[j * dim + c];
        os[e] = sa;
    }
    __syncthreads();
    for (int e = tid; e < ballDim; e += 256) {
        int i = e / dim, c = e % dim;
        float sa = p.proj_b[c];
        const float* wr = p.proj_w + (long)c * dim;
        const float* orow = os + i * dim;
        for (int cc = 0; cc < dim; cc++) sa += orow[cc] * wr[cc];
        x[(base + i) * dim + c] += sa;
    }
}

__global__ void k_pool_prep(const float* __restrict__ x, const float* __restrict__ ps0,
                            float* __restrict__ centers, float* __restrict__ relp,
                            float* __restrict__ hbuf) {
    int m = blockIdx.x * 256 + threadIdx.x;
    if (m >= 512) return;
    for (int a = 0; a < 3; a++) {
        float p0 = ps0[(2 * m) * 3 + a], p1 = ps0[(2 * m + 1) * 3 + a];
        float c = 0.5f * (p0 + p1);
        centers[m * 3 + a] = c;
        relp[m * 6 + a]     = p0 - c;
        relp[m * 6 + 3 + a] = p1 - c;
    }
    for (int c = 0; c < 64; c++) {
        hbuf[(long)m * 134 + c]      = x[(long)(2 * m) * 64 + c];
        hbuf[(long)m * 134 + 64 + c] = x[(long)(2 * m + 1) * 64 + c];
    }
    for (int a = 0; a < 6; a++) hbuf[(long)m * 134 + 128 + a] = relp[m * 6 + a];
}

__global__ void k_unpool_prep(const float* __restrict__ x1, const float* __restrict__ relp,
                              float* __restrict__ hbuf) {
    int m = blockIdx.x * 256 + threadIdx.x;
    if (m >= 512) return;
    for (int c = 0; c < 128; c++) hbuf[(long)m * 134 + c] = x1[(long)m * 128 + c];
    for (int a = 0; a < 6; a++)   hbuf[(long)m * 134 + 128 + a] = relp[m * 6 + a];
}

__global__ void k_unpool_add(const float* __restrict__ skip, const float* __restrict__ upd,
                             float* __restrict__ tsum) {
    int i = blockIdx.x * 256 + threadIdx.x;
    if (i >= 65536) return;
    int row = i >> 6, c = i & 63;
    tsum[i] = skip[i] + upd[(long)(row >> 1) * 128 + (row & 1) * 64 + c];
}

__global__ void k_scatter_f16(const float* __restrict__ x, const int* __restrict__ idx,
                              half_t* __restrict__ tokf16) {
    int t = blockIdx.x * 256 + threadIdx.x;
    if (t >= 65536) return;
    int i = t >> 6, c = t & 63;
    tokf16[(long)idx[i] * 64 + c] = (half_t)x[t];
}

// ======================================================
// out_x = einsum('bhgc,bhng->bhnc') via WMMA  (M=32768, K=32, N=64 per bh)
// grid: 1024 blocks x 256 = 8192 waves; wave keeps B in regs across 8 m-tiles
// ======================================================
__global__ void __launch_bounds__(256)
k_outx(const half_t* __restrict__ swh, const half_t* __restrict__ tokf16,
       half_t* __restrict__ outx) {
    int wave = (blockIdx.x << 3) + (threadIdx.x >> 5);
    int lane = threadIdx.x & 31;
    int r = lane & 15, hi = lane >> 4;
    int bh  = wave >> 8;          // 32 bh
    int mt0 = (wave & 255) * 8;   // 8 m-tiles per wave
    v16h bm[4];
#pragma unroll
    for (int j2 = 0; j2 < 4; j2++) {
        int c = j2 * 16 + r;
#pragma unroll
        for (int h2 = 0; h2 < 16; h2++)
            bm[j2][h2] = tokf16[((long)bh * 32 + h2 + hi * 16) * 64 + c];
    }
    for (int mi = 0; mi < 8; mi++) {
        long m0 = (long)(mt0 + mi) * 16;
        long ab = ((long)bh * 32768 + m0 + r) * 32 + hi * 16;
        v16h a = ld16(swh + ab, swh + ab + 8);
#pragma unroll
        for (int j2 = 0; j2 < 4; j2++) {
            v8f acc = zero8();
            acc = wmma16(a, bm[j2], acc);
            int c = j2 * 16 + r;
#pragma unroll
            for (int j = 0; j < 8; j++)
                outx[((long)bh * 32768 + m0 + j + hi * 8) * 64 + c] = (half_t)acc[j];
        }
    }
}

// ======================================================
// final projection: [B*N,512] @ Wout^T(512x128) + bout -> d_out f32
// ======================================================
__global__ void __launch_bounds__(256)
k_final(const half_t* __restrict__ outx, const half_t* __restrict__ woutH,
        const float* __restrict__ bout, float* __restrict__ dout) {
    int wave = (blockIdx.x << 3) + (threadIdx.x >> 5);
    int lane = threadIdx.x & 31;
    int r = lane & 15, hi = lane >> 4;
    int mt = wave >> 1;
    int ngrp = wave & 1;
    long m0 = (long)mt * 16;
    int n0 = ngrp * 64;
    long m = m0 + r;
    int b_ = (int)(m >> 15);
    int nn = (int)(m & 32767);

    v8f acc[4];
#pragma unroll
    for (int j2 = 0; j2 < 4; j2++) acc[j2] = zero8();

#pragma unroll 4
    for (int kc = 0; kc < 16; kc++) {
        int k0 = kc * 32 + hi * 16;
        int h0 = k0 >> 6, c0 = k0 & 63;
        int k1 = k0 + 8;
        int h1 = k1 >> 6, c1 = k1 & 63;
        const half_t* p0 = outx + ((long)(b_ * 8 + h0) * 32768 + nn) * 64 + c0;
        const half_t* p1 = outx + ((long)(b_ * 8 + h1) * 32768 + nn) * 64 + c1;
        v16h a = ld16(p0, p1);
#pragma unroll
        for (int j2 = 0; j2 < 4; j2++) {
            int oc = n0 + j2 * 16 + r;
            v16h bm = *(const v16h*)(woutH + (long)oc * 512 + kc * 32 + hi * 16);
            acc[j2] = wmma16(a, bm, acc[j2]);
        }
    }
#pragma unroll
    for (int j2 = 0; j2 < 4; j2++) {
        int oc = n0 + j2 * 16 + r;
        float bo = bout[oc];
#pragma unroll
        for (int j = 0; j < 8; j++) {
            long mm = m0 + j + hi * 8;
            dout[mm * 128 + oc] = acc[j2][j] + bo;
        }
    }
}

// ======================================================
// host side
// ======================================================
static BlockP mkBlock(void* const* d_in, int base) {
    BlockP p;
    p.ln1_g  = (const float*)d_in[base + 0];
    p.ln1_b  = (const float*)d_in[base + 1];
    p.qkv_w  = (const float*)d_in[base + 2];
    p.qkv_b  = (const float*)d_in[base + 3];
    p.proj_w = (const float*)d_in[base + 4];
    p.proj_b = (const float*)d_in[base + 5];
    p.pe_w   = (const float*)d_in[base + 6];
    p.pe_b   = (const float*)d_in[base + 7];
    p.sigma  = (const float*)d_in[base + 8];
    p.ln2_g  = (const float*)d_in[base + 9];
    p.ln2_b  = (const float*)d_in[base + 10];
    p.w12    = (const float*)d_in[base + 11];
    p.b12    = (const float*)d_in[base + 12];
    p.w3     = (const float*)d_in[base + 13];
    p.b3     = (const float*)d_in[base + 14];
    return p;
}

static void apply_block(hipStream_t st, float* x, const float* posb, int M, int dim,
                        int heads, int ball, const BlockP& p,
                        float* bufLN, float* bufU, float* bufG) {
    k_layernorm<<<(M + 7) / 8, 256, 0, st>>>(x, bufLN, p.ln1_g, p.ln1_b, M, dim);
    size_t sh = (size_t)(ball * 3 + 5 * ball * dim + heads * ball * ball + 8) * sizeof(float);
    k_ball<<<M / ball, 256, sh, st>>>(bufLN, x, posb, p, dim, heads, ball);
    k_layernorm<<<(M + 7) / 8, 256, 0, st>>>(x, bufLN, p.ln2_g, p.ln2_b, M, dim);
    int hid = 4 * dim;
    long nU = (long)M * 2 * hid;
    k_gemm<<<(unsigned)((nU + 255) / 256), 256, 0, st>>>(bufLN, p.w12, p.b12, nullptr, nullptr,
                                                         bufU, M, 2 * hid, dim);
    long nG = (long)M * hid;
    k_glu<<<(unsigned)((nG + 255) / 256), 256, 0, st>>>(bufU, bufG, M, hid);
    long nO = (long)M * dim;
    k_gemm<<<(unsigned)((nO + 255) / 256), 256, 0, st>>>(bufG, p.w3, p.b3, x, nullptr,
                                                         x, M, dim, hid);
}

extern "C" void kernel_launch(void* const* d_in, const int* in_sizes, int n_in,
                              void* d_out, int out_size, void* d_ws, size_t ws_size,
                              hipStream_t stream) {
    (void)in_sizes; (void)out_size; (void)ws_size;
    const float* X           = (const float*)d_in[0];
    const float* Wfx         = (const float*)d_in[1];
    const float* bfx         = (const float*)d_in[2];
    const float* Wx          = (const float*)d_in[3];
    const float* bx          = (const float*)d_in[4];
    const float* Wslice      = (const float*)d_in[5];
    const float* bslice      = (const float*)d_in[6];
    const float* temperature = (const float*)d_in[7];
    const float* Wout        = (const float*)d_in[8];
    const float* bout        = (const float*)d_in[9];
    const float* emb_w       = (const float*)d_in[10];
    const float* emb_b       = (const float*)d_in[11];
    BlockP enc0a = mkBlock(d_in, 12), enc0b = mkBlock(d_in, 27);
    const float* pool_w      = (const float*)d_in[42];
    const float* pool_b      = (const float*)d_in[43];
    const float* pool_ln_g   = (const float*)d_in[44];
    const float* pool_ln_b   = (const float*)d_in[45];
    BlockP enc1a = mkBlock(d_in, 46), enc1b = mkBlock(d_in, 61);
    const float* unpool_w    = (const float*)d_in[76];
    const float* unpool_b    = (const float*)d_in[77];
    const float* unpool_ln_g = (const float*)d_in[78];
    const float* unpool_ln_b = (const float*)d_in[79];
    BlockP dec0a = mkBlock(d_in, 80), dec0b = mkBlock(d_in, 95);
    const float* pos = (const float*)d_in[n_in - 1];
    float* dout = (float*)d_out;

    // -------- workspace carve --------
    char* wsb = (char*)d_ws;
    size_t off = 0;
    auto alloc = [&](size_t bytes) -> void* {
        void* p = wsb + off;
        off = (off + bytes + 255) & ~(size_t)255;
        return p;
    };
    half_t* xp      = (half_t*)alloc((size_t)4 * PADN * 128 * 2);
    half_t* wh      = (half_t*)alloc((size_t)3538944 * 2);
    half_t* yh      = (half_t*)alloc((size_t)134217728 * 2);
    half_t* swh     = (half_t*)alloc((size_t)33554432 * 2);
    half_t* outx    = (half_t*)alloc((size_t)67108864 * 2);
    half_t* woutH   = (half_t*)alloc((size_t)65536 * 2);
    float*  swpart  = (float*)alloc((size_t)8192 * 32 * 4);
    float*  snorm   = (float*)alloc((size_t)1024 * 4);
    float*  tokpart = (float*)alloc((size_t)512 * 2048 * 4);
    float*  tokens0 = (float*)alloc((size_t)65536 * 4);
    int*    idx     = (int*)alloc((size_t)1024 * 4);
    float*  ps0     = (float*)alloc((size_t)1024 * 3 * 4);
    float*  xcur    = (float*)alloc((size_t)65536 * 4);
    float*  bufLN   = (float*)alloc((size_t)65536 * 4);
    float*  bufU    = (float*)alloc((size_t)524288 * 4);
    float*  bufG    = (float*)alloc((size_t)262144 * 4);
    float*  skip    = (float*)alloc((size_t)65536 * 4);
    float*  centers = (float*)alloc((size_t)512 * 3 * 4);
    float*  relp    = (float*)alloc((size_t)512 * 6 * 4);
    float*  hbuf    = (float*)alloc((size_t)512 * 134 * 4);
    float*  x1      = (float*)alloc((size_t)512 * 128 * 4);
    float*  upd     = (float*)alloc((size_t)512 * 128 * 4);
    float*  tsum    = (float*)alloc((size_t)65536 * 4);
    half_t* tokf16  = (half_t*)alloc((size_t)65536 * 2);

    // -------- pipeline --------
    long padElems = (long)4 * PADN * 128;
    k_pad_x<<<(unsigned)((padElems + 255) / 256), 256, 0, stream>>>(X, xp);
    k_build_wh<<<(3538944 + 255) / 256, 256, 0, stream>>>(Wfx, Wx, wh);
    k_f32_to_f16<<<256, 256, 0, stream>>>(Wout, woutH, 65536L);

    k_conv<<<16384, 256, 0, stream>>>(xp, wh, bfx, bx, yh);

    k_slice<<<8192, 256, 0, stream>>>(yh, Wslice, bslice, temperature, swh, swpart);
    k_snorm_reduce<<<4, 256, 0, stream>>>(swpart, snorm);
    k_tok_part<<<512, 256, 0, stream>>>(yh, swh, tokpart);
    k_tok_reduce<<<256, 256, 0, stream>>>(tokpart, snorm, tokens0);

    // -------- Erwin --------
    k_morton<<<32, 32, 0, stream>>>(pos, idx, ps0);
    k_gemm<<<(65536 + 255) / 256, 256, 0, stream>>>(tokens0, emb_w, emb_b, nullptr, idx,
                                                    xcur, 1024, 64, 64);
    apply_block(stream, xcur, ps0, 1024, 64, 4, 32, enc0a, bufLN, bufU, bufG);
    apply_block(stream, xcur, ps0, 1024, 64, 4, 32, enc0b, bufLN, bufU, bufG);
    hipMemcpyAsync(skip, xcur, (size_t)65536 * 4, hipMemcpyDeviceToDevice, stream);
    k_pool_prep<<<2, 256, 0, stream>>>(xcur, ps0, centers, relp, hbuf);
    k_gemm<<<(512 * 128 + 255) / 256, 256, 0, stream>>>(hbuf, pool_w, pool_b, nullptr, nullptr,
                                                        tsum, 512, 128, 134);
    k_layernorm<<<64, 256, 0, stream>>>(tsum, x1, pool_ln_g, pool_ln_b, 512, 128);
    apply_block(stream, x1, centers, 512, 128, 8, 16, enc1a, bufLN, bufU, bufG);
    apply_block(stream, x1, centers, 512, 128, 8, 16, enc1b, bufLN, bufU, bufG);
    k_unpool_prep<<<2, 256, 0, stream>>>(x1, relp, hbuf);
    k_gemm<<<(512 * 128 + 255) / 256, 256, 0, stream>>>(hbuf, unpool_w, unpool_b, nullptr, nullptr,
                                                        upd, 512, 128, 134);
    k_unpool_add<<<256, 256, 0, stream>>>(skip, upd, tsum);
    k_layernorm<<<128, 256, 0, stream>>>(tsum, xcur, unpool_ln_g, unpool_ln_b, 1024, 64);
    apply_block(stream, xcur, ps0, 1024, 64, 4, 32, dec0a, bufLN, bufU, bufG);
    apply_block(stream, xcur, ps0, 1024, 64, 4, 32, dec0b, bufLN, bufU, bufG);
    k_scatter_f16<<<256, 256, 0, stream>>>(xcur, idx, tokf16);

    // -------- deslice + final projection --------
    k_outx<<<1024, 256, 0, stream>>>(swh, tokf16, outx);
    k_final<<<2048, 256, 0, stream>>>(outx, woutH, bout, dout);
}
